// DotProductAttention_25692494365250
// MI455X (gfx1250) — matrix operand order repaired and checked by reading, not compiled, not run on hardware
//
#include <hip/hip_runtime.h>
#include <hip/hip_bf16.h>

typedef __attribute__((ext_vector_type(16))) _Float16 v16h;
typedef __attribute__((ext_vector_type(8)))  _Float16 v8h;
typedef __attribute__((ext_vector_type(8)))  float    v8f;
typedef __attribute__((ext_vector_type(4)))  float    v4f;
typedef __attribute__((ext_vector_type(4)))  unsigned v4u;
typedef __attribute__((ext_vector_type(8)))  int      v8i;
typedef __attribute__((ext_vector_type(4)))  int      v4i;

// Problem constants (from reference): B=8, Q=2048, K=2048, D=512
#define BB 8
#define QQ 2048
#define KK 2048
#define DD 512

#define NE   ((size_t)BB * KK * DD)   // elements per matrix = 8388608
#define CHB  32768                    // bytes per 32-key chunk per matrix (f16)

// Pass-2 LDS map (dynamic, base offset 0):
//  K buf0 @ 0, K buf1 @ 34816, V buf0 @ 69632, V buf1 @ 104448, Psh @ 139264
// Each 32KB chunk grows to 34816B in LDS via TDM padding (+16B per 256B).
#define KB0 0u
#define KB1 34816u
#define VB0 69632u
#define VB1 104448u
#define PSH 139264u
#define SMEM_BYTES (139264 + 4096)

#define HAVE_TDM __has_builtin(__builtin_amdgcn_tensor_load_to_lds)

// ---------------------------------------------------------------------------
// Pass 1: f32 -> f16 conversion of K and V into WMMA-B-fragment chunk layout.
// K chunk layout: [c(16)][kt(2)][r (32)][slot (16)], n=r%16, d%32=16*(slot/8)+8*(r/16)+slot%8  (32KB, packed)
// V chunk layout: [nt=d/16 (32)][r (32)][slot (16)], d%16=r%16, key%32=16*(slot/8)+8*(r/16)+slot%8 (32KB, packed)
// ---------------------------------------------------------------------------
__global__ void __launch_bounds__(256)
convert_kv_kernel(const float* __restrict__ Kg, const float* __restrict__ Vg,
                  _Float16* __restrict__ Khf, _Float16* __restrict__ Vhf) {
  size_t g = (size_t)blockIdx.x * 256 + threadIdx.x;
  if (g < NE) {
    unsigned idx = (unsigned)g;
    int slot = idx & 15, r = (idx >> 4) & 31, kt = (idx >> 9) & 1;
    int c = (idx >> 10) & 15, kc = (idx >> 14) & 63, b = idx >> 20;
    int key = kc * 32 + kt * 16 + (r & 15);
    int d   = c * 32 + ((slot & 8) << 1) + ((r >> 4) << 3) + (slot & 7);
    Khf[idx] = (_Float16)Kg[((size_t)b * KK + key) * DD + d];
  } else {
    unsigned idx = (unsigned)(g - NE);
    int slot = idx & 15, r = (idx >> 4) & 31, nt = (idx >> 9) & 31;
    int kc = (idx >> 14) & 63, b = idx >> 20;
    int key = kc * 32 + ((slot & 8) << 1) + ((r >> 4) << 3) + (slot & 7);
    int d   = nt * 16 + (r & 15);
    Vhf[idx] = (_Float16)Vg[((size_t)b * KK + key) * DD + d];
  }
}

// ---------------------------------------------------------------------------
// TDM: DMA one packed 32KB chunk into LDS, injecting 16B pad per 256B so rows
// land at r*32 + (r>>3)*16 (bank-conflict-minimal for the 32B fragment loads).
// D#: count=1, type=2, data_size=8B, 1D tile of 4096 8B units,
// pad_interval=64DW(code 5), pad_amount=4DW(code 3).
// ---------------------------------------------------------------------------
__device__ __forceinline__ void tdm_load_chunk(unsigned long long ga, unsigned lds) {
#if HAVE_TDM
  v4u g0 = {1u, lds, (unsigned)ga, ((unsigned)(ga >> 32) & 0x01FFFFFFu) | 0x80000000u};
  v8i g1 = {(int)0x07530000,  // data_size=3(8B), pad_en, pad_interval=5, pad_amount=3
            (int)0x10000000,  // tensor_dim0[15:0]=4096 -> bits[31:16]
            (int)0x00010000,  // tensor_dim0 hi=0, tensor_dim1 lo=1
            (int)0x10000000,  // tensor_dim1 hi=0, tile_dim0=4096
            (int)0x00000001,  // tile_dim1=1, tile_dim2=0
            (int)0x00001000,  // tensor_dim0_stride lo = 4096
            (int)0x10000000,  // stride0 hi=0, tensor_dim1_stride lo=4096
            (int)0x00000000};
  v4i g2 = {0, 0, 0, 0}, g3 = {0, 0, 0, 0};
  v8i g4 = {0, 0, 0, 0, 0, 0, 0, 0};
  __builtin_amdgcn_tensor_load_to_lds(g0, g1, g2, g3, g4, 0);
#else
  (void)ga; (void)lds;
#endif
}

__device__ __forceinline__ void wait_tensor0() {
#if __has_builtin(__builtin_amdgcn_s_wait_tensorcnt)
  __builtin_amdgcn_s_wait_tensorcnt(0);
#else
  asm volatile("s_wait_tensorcnt 0x0" ::: "memory");
#endif
}

// ---------------------------------------------------------------------------
// Pass 2: flash attention. 4 waves/WG; wave = one 16-query tile; K/V chunks of
// 32 keys double-buffered in LDS via TDM; softmax faithful to reference
// (masked logits replaced with 0.0, softmax over ALL keys).
// ---------------------------------------------------------------------------
__global__ void __launch_bounds__(128, 1)
fa_wmma_kernel(const float* __restrict__ Qg, const int* __restrict__ VL,
               const _Float16* __restrict__ Khf, const _Float16* __restrict__ Vhf,
               float* __restrict__ Og) {
  extern __shared__ char smem[];
  _Float16* Psh = (_Float16*)(smem + PSH);

  const int tid  = threadIdx.x;
  const int wave = tid >> 5;
  const int lane = tid & 31;
  const int b    = blockIdx.y;
  const int q0   = blockIdx.x * 64 + wave * 16;
  const int vl   = VL[b];

  const unsigned long long kbase =
      (unsigned long long)(const void*)(Khf) + (unsigned long long)b * 64 * CHB;
  const unsigned long long vbase =
      (unsigned long long)(const void*)(Vhf) + (unsigned long long)b * 64 * CHB;

#if HAVE_TDM
  if (wave == 0) {           // prologue: DMA chunk 0 (overlaps Q load below)
    tdm_load_chunk(kbase, KB0);
    tdm_load_chunk(vbase, VB0);
  }
#endif

  // ---- Q tile -> registers, A-layout, pre-scaled by 1/sqrt(D) ----
  v16h qf[16];
  {
    const float qs = 0.044194173824159216f; // 1/sqrt(512)
    const int row = lane & 15;
    const int hi  = (lane >> 4) * 8;
    const v4f* qv = (const v4f*)(Qg + ((size_t)b * QQ + q0 + row) * DD);
    for (int c = 0; c < 16; ++c) {
      const int i0 = (c * 32 + hi) >> 2;     // v4f index of K {0..7 | 8..15}
      v4f a0 = qv[i0], a1 = qv[i0 + 1];      // 8 floats
      v4f b0 = qv[i0 + 4], b1 = qv[i0 + 5];  // +16 floats: K {16..23 | 24..31}
#pragma unroll
      for (int j = 0; j < 4; ++j) {
        qf[c][j]      = (_Float16)(a0[j] * qs);
        qf[c][4 + j]  = (_Float16)(a1[j] * qs);
        qf[c][8 + j]  = (_Float16)(b0[j] * qs);
        qf[c][12 + j] = (_Float16)(b1[j] * qs);
      }
    }
  }

  v8f acc[32];
#pragma unroll
  for (int i = 0; i < 32; ++i) acc[i] = (v8f){0.f,0.f,0.f,0.f,0.f,0.f,0.f,0.f};
  float m[8], l[8];
#pragma unroll
  for (int j = 0; j < 8; ++j) { m[j] = -1e30f; l[j] = 0.0f; }

  // swizzled per-lane fragment offset (f16 units): r*16 + (r>>3)*8
  const int roff = lane * 16 + ((lane >> 3) << 3);

  for (int kc = 0; kc < 64; ++kc) {
    const int cur = kc & 1;
#if HAVE_TDM
    if (wave == 0) wait_tensor0();       // chunk kc has landed
    __syncthreads();                     // release all waves; buf[nxt] is free
    if (wave == 0 && kc + 1 < 64) {      // prefetch chunk kc+1
      tdm_load_chunk(kbase + (unsigned long long)(kc + 1) * CHB, cur ? KB0 : KB1);
      tdm_load_chunk(vbase + (unsigned long long)(kc + 1) * CHB, cur ? VB0 : VB1);
    }
#else
    // fallback: cooperative f16 copy with software pad swizzle
    __syncthreads();
    {
      const char* kg = (const char*)(size_t)(kbase + (unsigned long long)kc * CHB);
      const char* vg = (const char*)(size_t)(vbase + (unsigned long long)kc * CHB);
      for (int rep = 0; rep < 16; ++rep) {
        const unsigned x = (rep * 128 + tid) * 16;      // byte offset in packed chunk
        const unsigned lx = x + ((x >> 8) << 4);        // +16B pad per 256B
        *(v8h*)(smem + (cur ? KB1 : KB0) + lx) = *(const v8h*)(kg + x);
        *(v8h*)(smem + (cur ? VB1 : VB0) + lx) = *(const v8h*)(vg + x);
      }
    }
    __syncthreads();
#endif
    const _Float16* Kb = (const _Float16*)(smem + (cur ? KB1 : KB0));
    const _Float16* Vb = (const _Float16*)(smem + (cur ? VB1 : VB0));

    // ---- S = Q * K^T, software-pipelined fragment loads (stride 544 f16) ----
    v8f s0 = (v8f){0.f,0.f,0.f,0.f,0.f,0.f,0.f,0.f};
    v8f s1 = (v8f){0.f,0.f,0.f,0.f,0.f,0.f,0.f,0.f};
    v16h ka = *(const v16h*)(Kb + 0 * 544 + roff);
    v16h kb = *(const v16h*)(Kb + 1 * 544 + roff);
#pragma unroll
    for (int c = 0; c < 16; ++c) {
      v16h kna = ka, knb = kb;
      if (c < 15) {
        kna = *(const v16h*)(Kb + ((c + 1) * 2 + 0) * 544 + roff);
        knb = *(const v16h*)(Kb + ((c + 1) * 2 + 1) * 544 + roff);
      }
      s0 = __builtin_amdgcn_wmma_f32_16x16x32_f16(false, qf[c], false, ka, (short)0, s0, false, false);
      s1 = __builtin_amdgcn_wmma_f32_16x16x32_f16(false, qf[c], false, kb, (short)0, s1, false, false);
      ka = kna; kb = knb;
    }

    // ---- mask (logit := 0.0 where key >= valid_len) + online softmax ----
    const int col = (kc << 5) + (lane & 15);
    const bool keep0 = col < vl;
    const bool keep1 = (col + 16) < vl;
    const int prow_lo = (lane < 16) ? 0 : 8;
    float corrv[8];
    bool mchanged = false;
#pragma unroll
    for (int j = 0; j < 8; ++j) {
      float a0 = keep0 ? s0[j] : 0.0f;
      float a1 = keep1 ? s1[j] : 0.0f;
      float t = fmaxf(a0, a1);
      t = fmaxf(t, __shfl_xor(t, 1, 16));
      t = fmaxf(t, __shfl_xor(t, 2, 16));
      t = fmaxf(t, __shfl_xor(t, 4, 16));
      t = fmaxf(t, __shfl_xor(t, 8, 16));
      const float mn = fmaxf(m[j], t);
      mchanged |= (mn > m[j]);
      const float corr = __expf(m[j] - mn);
      const float p0 = __expf(a0 - mn);
      const float p1 = __expf(a1 - mn);
      float rs = p0 + p1;
      rs += __shfl_xor(rs, 1, 16);
      rs += __shfl_xor(rs, 2, 16);
      rs += __shfl_xor(rs, 4, 16);
      rs += __shfl_xor(rs, 8, 16);
      l[j] = l[j] * corr + rs;
      m[j] = mn;
      corrv[j] = corr;
      _Float16* pp = Psh + wave * 512 + (j + prow_lo) * 32 + (lane & 15);
      pp[0]  = (_Float16)p0;
      pp[16] = (_Float16)p1;
    }
    // DS ops from one wave execute in order: the loads below observe the
    // stores above (incl. cross-lane) without a hardware wait. Compiler-level
    // barrier only, to prevent reordering.
    asm volatile("" ::: "memory");

    // reload P in A-layout (16x32)
    v16h pf;
    {
      const int off = (lane < 16) ? 0 : 8;
      const _Float16* pp = Psh + wave * 512 + (lane & 15) * 32;
      v8h plo = *(const v8h*)(pp + off);
      v8h phi = *(const v8h*)(pp + 16 + off);
#pragma unroll
      for (int j = 0; j < 8; ++j) { pf[j] = plo[j]; pf[8 + j] = phi[j]; }
    }

    // ---- rescale accumulators only when the running max moved (rare) ----
    if (__any(mchanged)) {
#pragma unroll
      for (int nt = 0; nt < 32; ++nt)
#pragma unroll
        for (int j = 0; j < 8; ++j) acc[nt][j] *= corrv[j];
    }

    // ---- O += P*V over 32 output-column tiles, pipelined V loads ----
    v16h vf = *(const v16h*)(Vb + 0 * 544 + roff);
#pragma unroll
    for (int nt = 0; nt < 32; ++nt) {
      v16h vn = vf;
      if (nt < 31) vn = *(const v16h*)(Vb + (nt + 1) * 544 + roff);
      acc[nt] = __builtin_amdgcn_wmma_f32_16x16x32_f16(false, pf, false, vf, (short)0, acc[nt], false, false);
      vf = vn;
    }
  }

  // ---- epilogue ----
  float linv[8];
#pragma unroll
  for (int j = 0; j < 8; ++j) linv[j] = 1.0f / l[j];
  float* Ob = Og + ((size_t)b * QQ + q0) * DD;
  const int rbase = (lane < 16) ? 0 : 8;
  const int ncol = lane & 15;
#pragma unroll
  for (int nt = 0; nt < 32; ++nt) {
#pragma unroll
    for (int j = 0; j < 8; ++j) {
      Ob[(size_t)(j + rbase) * DD + nt * 16 + ncol] = acc[nt][j] * linv[j];
    }
  }
}

extern "C" void kernel_launch(void* const* d_in, const int* in_sizes, int n_in,
                              void* d_out, int out_size, void* d_ws, size_t ws_size,
                              hipStream_t stream) {
  const float* q  = (const float*)d_in[0];
  const float* k  = (const float*)d_in[1];
  const float* v  = (const float*)d_in[2];
  const int*   vl = (const int*)d_in[3];
  float* out = (float*)d_out;
  (void)in_sizes; (void)n_in; (void)out_size; (void)ws_size;

  _Float16* khf = (_Float16*)d_ws;
  _Float16* vhf = khf + NE;

  // Pass 1: one-time f32->f16 repack of K and V (2*NE elements)
  convert_kv_kernel<<<dim3((unsigned)(2 * NE / 256)), dim3(256), 0, stream>>>(k, v, khf, vhf);

  // Pass 2: flash attention
  fa_wmma_kernel<<<dim3(QQ / 64, BB), dim3(128), SMEM_BYTES, stream>>>(q, vl, khf, vhf, out);
}
